// TopologyGuidedMKAN_66752381714565
// MI455X (gfx1250) — compile-verified, matrix operand order checked by reading, not run on hardware
//
#include <hip/hip_runtime.h>

// ---------------------------------------------------------------------------
// TopologyGuidedMKAN for MI455X (gfx1250, wave32, WMMA bf16 w/ f32 accum)
//
// B=8, T=1024, D=512, H=4 (hd=128), orders 2..6 (NUM_ORDERS=5)
//
// All matrix operands are staged ONCE as bf16 (halves fragment bytes, removes
// cvt from the WMMA inner loops; the whole intermediate working set ~200MB is
// ~L2-resident on the 192MB L2). Accumulation is f32 via
// v_wmma_f32_16x16x32_bf16.
//
// Pipeline:
//   0. bf16 staging of x, ctx, in_proj_w, out_proj_w, gate_w
//   1. qkv GEMMs (q from x, k/v from frequency_context) -> bf16 qkv   [WMMA]
//   2. flash attention per (b,h,16-row q block) -> bf16 attn          [WMMA]
//   3. out_proj + residual -> xa (f32)                                [WMMA]
//   4. selector: softmax(mean_t(xa) @ sel_w^T + sel_b) -> w[b,5]
//   5. ceff[b,n,d*7+k] = sum_i w[b,i]*cheb_i[d,n,k]  (bf16)
//   6. basis[b,t,d*7+k] = T_k(tanh(xa))              (bf16)
//   7. x1 = batched GEMM basis[b] @ ceff[b]^T                         [WMMA]
//   8. comb = x1 + depthwise_conv3(xa)   (f32 + bf16 copies)
//   9. out = x + comb * sigmoid(comb @ gate_w^T + gate_b)             [WMMA]
// ---------------------------------------------------------------------------

typedef __attribute__((ext_vector_type(16))) __bf16 v16bf;
typedef __attribute__((ext_vector_type(8)))  float  v8f;

#define BB   8
#define TT   1024
#define DD   512
#define HH   4
#define HD   128
#define BT   (BB * TT)        // 8192
#define KC   (DD * 7)         // 3584
#define QKVW (3 * DD)         // 1536

__device__ inline v8f wmma_bf16(v16bf a, v16bf b, v8f c) {
  return __builtin_amdgcn_wmma_f32_16x16x32_bf16(
      /*neg_a=*/false, a, /*neg_b=*/false, b,
      /*c_mod=*/(short)0, c, /*reuse_a=*/false, /*reuse_b=*/false);
}

// 16x32 bf16 A-fragment (also B-fragment for C = A @ W^T with W row-major
// [N,K]) from a bf16 row-major source. ISA 16-bit A layout (CDNA5 ISA):
// lanes 0-15: M=lane, K in [0,8) and [16,24); lanes 16-31: K in [8,16),[24,32)
// => per lane two contiguous 8-bf16 (16B) runs -> two global_load_b128.
__device__ inline v16bf load_fragA_bf(const __bf16* __restrict__ src, int ld) {
  const int lane = threadIdx.x & 31;
  const int r  = lane & 15;
  const int kh = (lane >> 4) << 3;       // 0 or 8
  const __bf16* p = src + (long long)r * ld + kh;
  v16bf a;
#pragma unroll
  for (int i = 0; i < 8; ++i) {
    a[i]     = p[i];
    a[8 + i] = p[i + 16];
  }
  return a;
}

// Same layout from LDS (leading dim 32). Used for P in attention.
__device__ inline v16bf load_fragA_lds(const __bf16* __restrict__ src) {
  const int lane = threadIdx.x & 31;
  const int r  = lane & 15;
  const int kh = (lane >> 4) << 3;
  v16bf a;
#pragma unroll
  for (int i = 0; i < 8; ++i) {
    a[i]     = src[r * 32 + kh + i];
    a[8 + i] = src[r * 32 + kh + 16 + i];
  }
  return a;
}

// 32(K) x 16(N) B-fragment from a K-major bf16 matrix (rows = K, ld = row
// stride). Used for V in attention (O += P @ V).
__device__ inline v16bf load_fragB_km_bf(const __bf16* __restrict__ src, int ld) {
  const int lane = threadIdx.x & 31;
  const int n  = lane & 15;
  const int kh = (lane >> 4) << 3;
  v16bf b;
#pragma unroll
  for (int i = 0; i < 8; ++i) {
    b[i]     = src[(long long)(kh + i) * ld + n];
    b[8 + i] = src[(long long)(kh + 16 + i) * ld + n];
  }
  return b;
}

// ---------------------------------------------------------------------------
// f32 -> bf16 staging copy
// ---------------------------------------------------------------------------
__global__ void cvt_bf16_kernel(const float* __restrict__ in,
                                __bf16* __restrict__ out, long long n) {
  const long long i = (long long)blockIdx.x * blockDim.x + threadIdx.x;
  if (i < n) out[i] = (__bf16)in[i];
}

// ---------------------------------------------------------------------------
// Generic batched WMMA GEMM: C[m,n] = A[m,:] . W[n,:] (+bias) (+epilogue)
// A, W are bf16. Block = 256 threads = 8 waves (4 M x 2 N); wave owns a 16x64
// C tile -> workgroup tile 64x128.
// EPI 0: C = acc (+bias)
// EPI 1: C = acc + bias + X[r,c]                       (residual)
// EPI 2: C = X[r,c] + X2[r,c] * sigmoid(acc + bias)    (gate)
// OUTBF: store C as bf16 instead of f32.
// ---------------------------------------------------------------------------
template <int EPI, bool OUTBF>
__global__ __launch_bounds__(256)
void wmma_gemm(const __bf16* __restrict__ A, long long sA,
               const __bf16* __restrict__ W, long long sW,
               const float* __restrict__ bias,
               void* __restrict__ Cv, long long sC, int ldc,
               const float* __restrict__ X, const float* __restrict__ X2,
               int ldx, int M, int N, int K) {
  const int bz = blockIdx.z;
  A += (long long)bz * sA;
  W += (long long)bz * sW;

  const int wave = threadIdx.x >> 5;
  const int wy = wave & 3, wx = wave >> 2;
  const int m0 = blockIdx.y * 64 + wy * 16;
  const int n0 = blockIdx.x * 128 + wx * 64;
  if (m0 >= M || n0 >= N) return;   // uniform per wave; tiles are exact anyway

  const __bf16* Arow = A + (long long)m0 * K;
  v8f acc[4] = {};

  for (int k0 = 0; k0 < K; k0 += 32) {
    v16bf af = load_fragA_bf(Arow + k0, K);
    if (k0 + 64 < K) {               // hint next A tile (global_prefetch_b8)
      __builtin_prefetch(Arow + k0 + 64, 0, 1);
    }
#pragma unroll
    for (int j = 0; j < 4; ++j) {
      v16bf bf = load_fragA_bf(W + (long long)(n0 + j * 16) * K + k0, K);
      acc[j] = wmma_bf16(af, bf, acc[j]);
    }
  }

  // C fragment layout: VGPR i -> rows rb+i, lane&15 -> column
  const int lane = threadIdx.x & 31;
  const int col  = lane & 15;
  const int rb   = (lane >> 4) << 3;
#pragma unroll
  for (int j = 0; j < 4; ++j) {
#pragma unroll
    for (int i = 0; i < 8; ++i) {
      const int r = m0 + rb + i;
      const int c = n0 + j * 16 + col;
      float v = acc[j][i];
      if (bias) v += bias[c];
      if (EPI == 1) {
        v += X[(long long)r * ldx + c];
      } else if (EPI == 2) {
        const float g = 1.0f / (1.0f + __expf(-v));
        v = X[(long long)r * ldx + c] + X2[(long long)r * ldx + c] * g;
      }
      if (OUTBF) {
        ((__bf16*)Cv)[(long long)bz * sC + (long long)r * ldc + c] = (__bf16)v;
      } else {
        ((float*)Cv)[(long long)bz * sC + (long long)r * ldc + c] = v;
      }
    }
  }
}

// ---------------------------------------------------------------------------
// Flash attention: one wave per (b, h, 16-row q block). Online softmax, scores
// and probabilities kept in WMMA C layout; P reshaped to an A-fragment through
// a 16x32 bf16 LDS tile.  qkv layout: bf16 [BT, 1536] = [q | k | v].
// Writes attn output as bf16 (consumed by the out_proj GEMM).
// ---------------------------------------------------------------------------
__global__ __launch_bounds__(32)
void flash_attn_kernel(const __bf16* __restrict__ qkv,
                       __bf16* __restrict__ attn_out) {
  const int qb = blockIdx.x;   // 0..63 (T/16)
  const int h  = blockIdx.y;   // 0..3
  const int b  = blockIdx.z;   // 0..7
  const int lane = threadIdx.x;
  const int col  = lane & 15;
  const int rb   = (lane >> 4) << 3;
  const float scale = 0.08838834764831845f;   // 1/sqrt(128)

  const long long rowQ0 = (long long)b * TT + qb * 16;
  const __bf16* qbase = qkv + rowQ0 * QKVW + h * HD;

  v16bf Qf[4];
#pragma unroll
  for (int c = 0; c < 4; ++c) Qf[c] = load_fragA_bf(qbase + c * 32, QKVW);

  v8f O[8] = {};
  float m[8], l[8];
#pragma unroll
  for (int e = 0; e < 8; ++e) { m[e] = -3.0e38f; l[e] = 0.0f; }

  __shared__ __bf16 P[16 * 32];

  for (int kb = 0; kb < TT; kb += 32) {
    const __bf16* kbase0 = qkv + ((long long)b * TT + kb) * QKVW + DD + h * HD;
    const __bf16* kbase1 = kbase0 + 16 * QKVW;

    v8f S0 = {}, S1 = {};
#pragma unroll
    for (int c = 0; c < 4; ++c) {
      S0 = wmma_bf16(Qf[c], load_fragA_bf(kbase0 + c * 32, QKVW), S0);
      S1 = wmma_bf16(Qf[c], load_fragA_bf(kbase1 + c * 32, QKVW), S1);
    }

    __syncthreads();  // WAR: previous iteration's P reads finished

#pragma unroll
    for (int e = 0; e < 8; ++e) {
      const float s0 = S0[e] * scale;
      const float s1 = S1[e] * scale;
      float mx = fmaxf(s0, s1);
#pragma unroll
      for (int off = 8; off; off >>= 1) mx = fmaxf(mx, __shfl_xor(mx, off, 16));
      const float mn = fmaxf(m[e], mx);
      const float alpha = __expf(m[e] - mn);
      const float p0 = __expf(s0 - mn);
      const float p1 = __expf(s1 - mn);
      float rs = p0 + p1;
#pragma unroll
      for (int off = 8; off; off >>= 1) rs += __shfl_xor(rs, off, 16);
      l[e] = l[e] * alpha + rs;
      m[e] = mn;
#pragma unroll
      for (int f = 0; f < 8; ++f) O[f][e] *= alpha;
      P[(rb + e) * 32 + col]      = (__bf16)p0;
      P[(rb + e) * 32 + 16 + col] = (__bf16)p1;
    }

    __syncthreads();  // RAW: P visible before re-load as A fragment

    const v16bf Pf = load_fragA_lds(P);
    const __bf16* vbase =
        qkv + ((long long)b * TT + kb) * QKVW + 2 * DD + h * HD;
#pragma unroll
    for (int f = 0; f < 8; ++f) {
      O[f] = wmma_bf16(Pf, load_fragB_km_bf(vbase + f * 16, QKVW), O[f]);
    }
  }

  __bf16* obase = attn_out + rowQ0 * DD + h * HD;
#pragma unroll
  for (int e = 0; e < 8; ++e) {
    const float inv = 1.0f / l[e];
#pragma unroll
    for (int f = 0; f < 8; ++f) {
      obase[(long long)(rb + e) * DD + f * 16 + col] =
          (__bf16)(O[f][e] * inv);
    }
  }
}

// ---------------------------------------------------------------------------
// Order selector: w[b,:] = softmax(mean_t(xa[b]) @ sel_w^T + sel_b)
// ---------------------------------------------------------------------------
__global__ __launch_bounds__(256)
void selector_kernel(const float* __restrict__ xa,
                     const float* __restrict__ sel_w,
                     const float* __restrict__ sel_b,
                     float* __restrict__ wsel) {
  const int b = blockIdx.x;
  const int tid = threadIdx.x;
  float part[5] = {0.f, 0.f, 0.f, 0.f, 0.f};
  for (int d = tid; d < DD; d += 256) {
    const float* p = xa + (long long)b * TT * DD + d;
    float s = 0.f;
    for (int t = 0; t < TT; ++t) s += p[(long long)t * DD];
    s *= (1.0f / (float)TT);
#pragma unroll
    for (int o = 0; o < 5; ++o) part[o] += s * sel_w[o * DD + d];
  }
  __shared__ float red[5][256];
#pragma unroll
  for (int o = 0; o < 5; ++o) red[o][tid] = part[o];
  __syncthreads();
  for (int st = 128; st; st >>= 1) {
    if (tid < st) {
#pragma unroll
      for (int o = 0; o < 5; ++o) red[o][tid] += red[o][tid + st];
    }
    __syncthreads();
  }
  if (tid == 0) {
    float lg[5], mx = -3.0e38f;
#pragma unroll
    for (int o = 0; o < 5; ++o) { lg[o] = red[o][0] + sel_b[o]; mx = fmaxf(mx, lg[o]); }
    float sm = 0.f;
#pragma unroll
    for (int o = 0; o < 5; ++o) { lg[o] = __expf(lg[o] - mx); sm += lg[o]; }
    const float inv = 1.0f / sm;
#pragma unroll
    for (int o = 0; o < 5; ++o) wsel[b * 5 + o] = lg[o] * inv;
  }
}

// ---------------------------------------------------------------------------
// ceff[b, n, d*7+k] = sum_i w[b,i] * cheb_i[d, n, k]  (k past order -> 0)
// Stored bf16 (B operand of the Chebyshev GEMM).
// ---------------------------------------------------------------------------
__global__ void build_ceff_kernel(const float* __restrict__ w,
                                  const float* __restrict__ c2,
                                  const float* __restrict__ c3,
                                  const float* __restrict__ c4,
                                  const float* __restrict__ c5,
                                  const float* __restrict__ c6,
                                  __bf16* __restrict__ ceff) {
  const long long i = (long long)blockIdx.x * blockDim.x + threadIdx.x;
  const long long total = (long long)BB * DD * KC;
  if (i >= total) return;
  const int k = (int)(i % 7);
  long long t = i / 7;
  const int d = (int)(t % DD);
  t /= DD;
  const int n = (int)(t % DD);
  const int b = (int)(t / DD);
  const long long dn = (long long)d * DD + n;
  float acc = 0.f;
  if (k < 3) acc += w[b * 5 + 0] * c2[dn * 3 + k];
  if (k < 4) acc += w[b * 5 + 1] * c3[dn * 4 + k];
  if (k < 5) acc += w[b * 5 + 2] * c4[dn * 5 + k];
  if (k < 6) acc += w[b * 5 + 3] * c5[dn * 6 + k];
  acc += w[b * 5 + 4] * c6[dn * 7 + k];
  ceff[i] = (__bf16)acc;
}

// ---------------------------------------------------------------------------
// basis[bt, d*7+k] = T_k(tanh(xa[bt,d])), k = 0..6   (bf16)
// ---------------------------------------------------------------------------
__global__ void build_basis_kernel(const float* __restrict__ xa,
                                   __bf16* __restrict__ basis) {
  const long long i = (long long)blockIdx.x * blockDim.x + threadIdx.x;
  if (i >= (long long)BT * DD) return;
  const int d = (int)(i & (DD - 1));
  const long long bt = i >> 9;
  const float t = tanhf(xa[i]);
  __bf16* o = basis + bt * KC + (long long)d * 7;
  float a = 1.f, bcur = t;
  o[0] = (__bf16)1.0f;
  o[1] = (__bf16)t;
#pragma unroll
  for (int k = 2; k < 7; ++k) {
    const float c = 2.0f * t * bcur - a;
    o[k] = (__bf16)c;
    a = bcur;
    bcur = c;
  }
}

// ---------------------------------------------------------------------------
// comb = x1 + depthwise_conv3(xa): write f32 (for gate epilogue) + bf16 (GEMM A)
// ---------------------------------------------------------------------------
__global__ void conv_combine_kernel(const float* __restrict__ xa,
                                    const float* __restrict__ x1,
                                    const float* __restrict__ conv_w,
                                    float* __restrict__ comb_f,
                                    __bf16* __restrict__ comb_bf) {
  const long long i = (long long)blockIdx.x * blockDim.x + threadIdx.x;
  if (i >= (long long)BT * DD) return;
  const int d = (int)(i & (DD - 1));
  const long long bt = i >> 9;
  const int t = (int)(bt & (TT - 1));
  const float cur = xa[i];
  const float prv = (t > 0)      ? xa[i - DD] : 0.f;
  const float nxt = (t < TT - 1) ? xa[i + DD] : 0.f;
  const float x2 = conv_w[d * 3 + 0] * prv +
                   conv_w[d * 3 + 1] * cur +
                   conv_w[d * 3 + 2] * nxt;
  const float v = x1[i] + x2;
  comb_f[i]  = v;
  comb_bf[i] = (__bf16)v;
}

// ---------------------------------------------------------------------------
// Host launcher
// ---------------------------------------------------------------------------
extern "C" void kernel_launch(void* const* d_in, const int* in_sizes, int n_in,
                              void* d_out, int out_size, void* d_ws, size_t ws_size,
                              hipStream_t stream) {
  const float* x     = (const float*)d_in[0];
  const float* ctx   = (const float*)d_in[1];
  const float* in_w  = (const float*)d_in[2];   // [1536, 512]
  const float* in_b  = (const float*)d_in[3];
  const float* out_w = (const float*)d_in[4];   // [512, 512]
  const float* out_b = (const float*)d_in[5];
  const float* sel_w = (const float*)d_in[6];   // [5, 512]
  const float* sel_b = (const float*)d_in[7];
  const float* cv_w  = (const float*)d_in[8];   // [512, 1, 3]
  const float* gw    = (const float*)d_in[9];   // [512, 512]
  const float* gb    = (const float*)d_in[10];
  const float* c2    = (const float*)d_in[11];
  const float* c3    = (const float*)d_in[12];
  const float* c4    = (const float*)d_in[13];
  const float* c5    = (const float*)d_in[14];
  const float* c6    = (const float*)d_in[15];
  float* out = (float*)d_out;

  // ---- workspace carve-up (16B aligned chunks) ----
  char* cur = (char*)d_ws;
  auto takeF = [&](long long n) { float*  p = (float*)cur;  cur += n * 4; return p; };
  auto takeB = [&](long long n) { __bf16* p = (__bf16*)cur; cur += ((n * 2 + 15) & ~15LL); return p; };

  __bf16* xbf     = takeB((long long)BT * DD);        //  8192x512
  __bf16* ctxbf   = takeB((long long)BT * DD);
  __bf16* wqkv_bf = takeB((long long)QKVW * DD);      //  1536x512
  __bf16* wout_bf = takeB((long long)DD * DD);
  __bf16* wgate_bf= takeB((long long)DD * DD);
  __bf16* qkv     = takeB((long long)BT * QKVW);      //  8192x1536
  __bf16* attn    = takeB((long long)BT * DD);
  __bf16* ceff    = takeB((long long)BB * DD * KC);   //  8x512x3584
  __bf16* basis   = takeB((long long)BT * KC);        //  8192x3584
  __bf16* comb_bf = takeB((long long)BT * DD);
  float*  xa      = takeF((long long)BT * DD);
  float*  x1      = takeF((long long)BT * DD);
  float*  comb_f  = takeF((long long)BT * DD);
  float*  wsel    = takeF(64);

  const dim3 blk(256);
  auto cvt = [&](const float* src, __bf16* dst, long long n) {
    cvt_bf16_kernel<<<(unsigned)((n + 255) / 256), 256, 0, stream>>>(src, dst, n);
  };

  // 0) bf16 staging (one pass each; tiny vs the matmul work)
  cvt(x,     xbf,      (long long)BT * DD);
  cvt(ctx,   ctxbf,    (long long)BT * DD);
  cvt(in_w,  wqkv_bf,  (long long)QKVW * DD);
  cvt(out_w, wout_bf,  (long long)DD * DD);
  cvt(gw,    wgate_bf, (long long)DD * DD);

  // 1) q = x @ Wq^T + bq ; [k|v] = ctx @ Wkv^T + bkv  (bf16 packed qkv)
  wmma_gemm<0, true><<<dim3(DD / 128, BT / 64, 1), blk, 0, stream>>>(
      xbf, 0, wqkv_bf, 0, in_b, qkv, 0, QKVW, nullptr, nullptr, 0,
      BT, DD, DD);
  wmma_gemm<0, true><<<dim3(2 * DD / 128, BT / 64, 1), blk, 0, stream>>>(
      ctxbf, 0, wqkv_bf + (long long)DD * DD, 0, in_b + DD,
      qkv + DD, 0, QKVW, nullptr, nullptr, 0, BT, 2 * DD, DD);

  // 2) flash attention
  flash_attn_kernel<<<dim3(TT / 16, HH, BB), 32, 0, stream>>>(qkv, attn);

  // 3) xa = x + attn @ out_w^T + out_b
  wmma_gemm<1, false><<<dim3(DD / 128, BT / 64, 1), blk, 0, stream>>>(
      attn, 0, wout_bf, 0, out_b, xa, 0, DD, x, nullptr, DD, BT, DD, DD);

  // 4) routing weights
  selector_kernel<<<BB, 256, 0, stream>>>(xa, sel_w, sel_b, wsel);

  // 5) effective coefficients (bf16)
  build_ceff_kernel<<<(unsigned)(((long long)BB * DD * KC + 255) / 256), 256, 0,
                      stream>>>(wsel, c2, c3, c4, c5, c6, ceff);

  // 6) Chebyshev basis (bf16)
  build_basis_kernel<<<(unsigned)(((long long)BT * DD + 255) / 256), 256, 0,
                       stream>>>(xa, basis);

  // 7) x1[b] = basis[b] @ ceff[b]^T   (batched over B)
  wmma_gemm<0, false><<<dim3(DD / 128, TT / 64, BB), blk, 0, stream>>>(
      basis, (long long)TT * KC, ceff, (long long)DD * KC, nullptr,
      x1, (long long)TT * DD, DD, nullptr, nullptr, 0, TT, DD, KC);

  // 8) comb = x1 + depthwise conv  (f32 + bf16)
  conv_combine_kernel<<<(unsigned)(((long long)BT * DD + 255) / 256), 256, 0,
                        stream>>>(xa, x1, cv_w, comb_f, comb_bf);

  // 9) out = x + comb * sigmoid(comb @ gw^T + gb)
  wmma_gemm<2, false><<<dim3(DD / 128, BT / 64, 1), blk, 0, stream>>>(
      comb_bf, 0, wgate_bf, 0, gb, out, 0, DD, x, comb_f, DD, BT, DD, DD);
}
